// CLCRec_21251498181534
// MI455X (gfx1250) — compile-verified, matrix operand order checked by reading, not run on hardware
//
#include <hip/hip_runtime.h>
#include <hip/hip_bf16.h>
#include <math.h>

typedef __attribute__((ext_vector_type(16))) _Float16 v16h;
typedef __attribute__((ext_vector_type(8)))  float    v8f;

#define NUM_USER  200000
#define NUM_ITEM  200000
#define DIM_E     64
#define DIM_FEAT  128
#define BATCH     16384
#define GRP       17
#define NTOT      (BATCH * GRP)     /* 278528 */
#define NREPL     139264            /* int(NTOT * 0.5) */
#define TEMPR     0.2f

/* Use CDNA5 async global->LDS copies (ASYNCcnt) for weight staging. */
#define USE_ASYNC_STAGE 1

/* ---------------- helpers ---------------- */

union FragU { v16h v; uint4 q[2]; };

/* Load a 16x32 f16 WMMA fragment from an LDS tile stored row-major
   ([rows][K], stride in halves). Works for A ([M][K]) and for B when the
   tile holds B^T ([N][K]).  Per ISA 7.12.2 16-bit A layout:
   lanes 0-15 : row = lane,    K = k0+0..7  and k0+16..23
   lanes 16-31: row = lane-16, K = k0+8..15 and k0+24..31           */
__device__ __forceinline__ v16h load_frag(const _Float16* tile, int row0,
                                          int k0, int stride, int lane) {
  int hl = lane & 15, hi = lane >> 4;
  const _Float16* p = tile + (row0 + hl) * stride + k0 + hi * 8;
  FragU f;
  f.q[0] = *(const uint4*)(p);        /* 8 halves  */
  f.q[1] = *(const uint4*)(p + 16);   /* 8 halves, +16 K */
  return f.v;
}

__device__ __forceinline__ float wave_sum(float v) {
  #pragma unroll
  for (int m = 16; m >= 1; m >>= 1) v += __shfl_xor(v, m, 32);
  return v;
}

/* ---------------- weight pre-conversion (f32 -> f16, transposed) -------- */

__global__ void convert_weights(const float* __restrict__ W1,
                                const float* __restrict__ W2,
                                _Float16* __restrict__ W1t,
                                _Float16* __restrict__ W2t) {
  int i = blockIdx.x * blockDim.x + threadIdx.x;
  if (i < DIM_FEAT * 256) {                 /* W1: [128][256] -> W1t[n][k] */
    int k = i / 256, n = i % 256;
    W1t[n * DIM_FEAT + k] = (_Float16)W1[i];
  }
  int j = i - DIM_FEAT * 256;
  if (j >= 0 && j < 256 * DIM_E) {          /* W2: [256][64] -> W2t[n][k] */
    int k = j / DIM_E, n = j % DIM_E;
    W2t[n * 256 + k] = (_Float16)W2[j];
  }
}

/* ---------------- replacement mask ---------------- */

__global__ void zero_mask(int* __restrict__ mask) {
  int i = blockIdx.x * 256 + threadIdx.x;
  if (i < NTOT) mask[i] = 0;
}
__global__ void set_mask(const int* __restrict__ rand_index,
                         int* __restrict__ mask) {
  int i = blockIdx.x * 256 + threadIdx.x;
  if (i < NREPL) mask[rand_index[i]] = 1;   /* idempotent */
}

/* ---------------- fused encoder: feature = (lrelu(l2n(V)@W1+b1))@W2+b2 -- */

/* dynamic-LDS byte offsets (no static __shared__ in this kernel) */
#define OFF_W1 0u          /* 256x128 f16 = 65536 B */
#define OFF_W2 65536u      /*  64x256 f16 = 32768 B */
#define OFF_A  98304u      /*  64x128 f16 = 16384 B */
#define OFF_H  114688u     /*  64x256 f16 = 32768 B */
#define OFF_B1 147456u     /*  256 f32   =  1024 B */
#define OFF_B2 148480u     /*   64 f32   =   256 B */
#define LDS_BYTES 148736u

__global__ __launch_bounds__(128) void encoder_kernel(
    const float* __restrict__ v_feat, const _Float16* __restrict__ W1t,
    const _Float16* __restrict__ W2t, const float* __restrict__ b1,
    const float* __restrict__ b2, float* __restrict__ feature) {
  extern __shared__ __align__(16) char smem[];
  _Float16* sW1 = (_Float16*)(smem + OFF_W1);
  _Float16* sW2 = (_Float16*)(smem + OFF_W2);
  _Float16* sA  = (_Float16*)(smem + OFF_A);
  _Float16* sH  = (_Float16*)(smem + OFF_H);
  float*    sB1 = (float*)(smem + OFF_B1);
  float*    sB2 = (float*)(smem + OFF_B2);

  const int tid  = threadIdx.x;
  const int lane = tid & 31;
  const int wave = tid >> 5;
  const int row0 = blockIdx.x * 64;

  /* ---- stage weights + biases into LDS ---- */
#if USE_ASYNC_STAGE
  {
    const uint4* g1 = (const uint4*)W1t;
    for (int i = tid; i < 4096; i += 128) {        /* 65536 B */
      unsigned ldsa = OFF_W1 + (unsigned)i * 16u;
      const uint4* gp = g1 + i;
      asm volatile("global_load_async_to_lds_b128 %0, %1, off"
                   :: "v"(ldsa), "v"(gp) : "memory");
    }
    const uint4* g2 = (const uint4*)W2t;
    for (int i = tid; i < 2048; i += 128) {        /* 32768 B */
      unsigned ldsa = OFF_W2 + (unsigned)i * 16u;
      const uint4* gp = g2 + i;
      asm volatile("global_load_async_to_lds_b128 %0, %1, off"
                   :: "v"(ldsa), "v"(gp) : "memory");
    }
  }
#else
  {
    const uint4* g1 = (const uint4*)W1t; uint4* s1 = (uint4*)sW1;
    for (int i = tid; i < 4096; i += 128) s1[i] = g1[i];
    const uint4* g2 = (const uint4*)W2t; uint4* s2 = (uint4*)sW2;
    for (int i = tid; i < 2048; i += 128) s2[i] = g2[i];
  }
#endif
  for (int i = tid; i < 256; i += 128) sB1[i] = b1[i];
  if (tid < 64) sB2[tid] = b2[tid];

  /* ---- l2-normalize 16 rows per wave, write f16 A tile (overlaps async) */
  #pragma unroll 1
  for (int it = 0; it < 16; ++it) {
    int r = wave * 16 + it;
    const float4 x = *(const float4*)(v_feat + (size_t)(row0 + r) * DIM_FEAT + lane * 4);
    float ss = wave_sum(x.x * x.x + x.y * x.y + x.z * x.z + x.w * x.w);
    float inv = 1.0f / fmaxf(sqrtf(ss), 1e-12f);
    _Float16* d = sA + r * DIM_FEAT + lane * 4;
    d[0] = (_Float16)(x.x * inv); d[1] = (_Float16)(x.y * inv);
    d[2] = (_Float16)(x.z * inv); d[3] = (_Float16)(x.w * inv);
  }
#if USE_ASYNC_STAGE
  asm volatile("s_wait_asynccnt 0x0" ::: "memory");
#endif
  __syncthreads();

  const int n  = lane & 15;
  const int hi = lane >> 4;

  /* ---- H = leaky_relu(A @ W1 + b1)  (per wave: 16 rows x 256 cols) ---- */
  {
    v16h aA[4];                                   /* A frags hoisted: K=128 */
    #pragma unroll
    for (int kb = 0; kb < 4; ++kb)
      aA[kb] = load_frag(sA, wave * 16, kb * 32, DIM_FEAT, lane);

    #pragma unroll 1
    for (int nb = 0; nb < 16; ++nb) {
      v8f acc = {};
      #pragma unroll
      for (int kb = 0; kb < 4; ++kb) {
        v16h b = load_frag(sW1, nb * 16, kb * 32, DIM_FEAT, lane);
        acc = __builtin_amdgcn_wmma_f32_16x16x32_f16(false, aA[kb], false, b,
                                                     (short)0, acc, false, false);
      }
      #pragma unroll
      for (int i = 0; i < 8; ++i) {
        int m = hi * 8 + i;
        float h = acc[i] + sB1[nb * 16 + n];
        h = fmaxf(h, 0.01f * h);                  /* leaky_relu, slope 0.01 */
        sH[(wave * 16 + m) * 256 + nb * 16 + n] = (_Float16)h;
      }
    }
  }
  __syncthreads();

  /* ---- feature = H @ W2 + b2  (per wave: 16 rows x 64 cols) ---- */
  {
    v16h aH[8];                                   /* A frags hoisted: K=256 */
    #pragma unroll
    for (int kb = 0; kb < 8; ++kb)
      aH[kb] = load_frag(sH, wave * 16, kb * 32, 256, lane);

    #pragma unroll 1
    for (int nb = 0; nb < 4; ++nb) {
      v8f acc = {};
      #pragma unroll
      for (int kb = 0; kb < 8; ++kb) {
        v16h b = load_frag(sW2, nb * 16, kb * 32, 256, lane);
        acc = __builtin_amdgcn_wmma_f32_16x16x32_f16(false, aH[kb], false, b,
                                                     (short)0, acc, false, false);
      }
      #pragma unroll
      for (int i = 0; i < 8; ++i) {
        int m = hi * 8 + i;
        feature[(size_t)(row0 + wave * 16 + m) * DIM_E + nb * 16 + n] =
            acc[i] + sB2[nb * 16 + n];
      }
    }
  }
}

/* ---------------- fused dual contrastive loss ---------------- */

__global__ __launch_bounds__(256) void loss_kernel(
    const float* __restrict__ feature, const float* __restrict__ id_emb,
    const int* __restrict__ user_t, const int* __restrict__ item_t,
    const int* __restrict__ mask, float* __restrict__ partial) {
  __shared__ float acc_s[8];
  const int tid = threadIdx.x, lane = tid & 31, wave = tid >> 5;
  const int b = blockIdx.x * 8 + wave;   /* one group per wave */
  const int d = lane * 2;                /* 2 dims per lane, 64 total */

  /* anchor1 = l2norm(id_emb[pos_item_b]) — constant within group */
  int pos_item = item_t[b * GRP];
  float2 pe = *(const float2*)(id_emb + (size_t)pos_item * DIM_E + d);
  float invp = 1.0f / fmaxf(sqrtf(wave_sum(pe.x * pe.x + pe.y * pe.y)), 1e-12f);
  float a1x = pe.x * invp, a1y = pe.y * invp;

  float tot1 = 0.f, pos1 = 0.f, tot2 = 0.f, pos2 = 0.f;
  #pragma unroll 1
  for (int g = 0; g < GRP; ++g) {
    int idx = b * GRP + g;
    int iid = item_t[idx];
    int ii  = iid - NUM_USER;
    ii = ii < 0 ? 0 : (ii > NUM_ITEM - 1 ? NUM_ITEM - 1 : ii);
    float2 f  = *(const float2*)(feature + (size_t)ii * DIM_E + d);
    int   u   = user_t[idx];
    float2 ue = *(const float2*)(id_emb + (size_t)u * DIM_E + d);
    float2 ie = *(const float2*)(id_emb + (size_t)iid * DIM_E + d);
    int   mk  = mask[idx];
    float avx = mk ? f.x : ie.x;
    float avy = mk ? f.y : ie.y;

    float sff = f.x * f.x + f.y * f.y;     /* |feat|^2     */
    float saf = a1x * f.x + a1y * f.y;     /* anchor1.feat */
    float sua = ue.x * avx + ue.y * avy;   /* user.allv    */
    #pragma unroll
    for (int mm = 16; mm >= 1; mm >>= 1) {
      sff += __shfl_xor(sff, mm, 32);
      saf += __shfl_xor(saf, mm, 32);
      sua += __shfl_xor(sua, mm, 32);
    }
    float invf = 1.0f / fmaxf(sqrtf(sff), 1e-12f);
    float s1 = expf((saf * invf) * (1.0f / TEMPR));
    float s2 = expf(sua * (1.0f / TEMPR));
    tot1 += s1; tot2 += s2;
    if (g == 0) { pos1 = s1; pos2 = s2; }
  }
  if (lane == 0) {
    float c1 = -logf(pos1 / (tot1 + 1e-8f) + 1e-8f);
    float c2 = -logf(pos2 / (tot2 + 1e-8f) + 1e-8f);
    acc_s[wave] = 0.5f * c1 + 0.5f * c2;   /* LR_LAMBDA = 0.5 */
  }
  __syncthreads();
  if (tid == 0) {
    float s = 0.f;
    #pragma unroll
    for (int w = 0; w < 8; ++w) s += acc_s[w];
    partial[blockIdx.x] = s;
  }
}

__global__ void final_reduce(const float* __restrict__ partial,
                             float* __restrict__ out) {
  __shared__ float s[256];
  float a = 0.f;
  for (int i = threadIdx.x; i < 2048; i += 256) a += partial[i];
  s[threadIdx.x] = a;
  __syncthreads();
  for (int st = 128; st >= 1; st >>= 1) {
    if (threadIdx.x < st) s[threadIdx.x] += s[threadIdx.x + st];
    __syncthreads();
  }
  if (threadIdx.x == 0) out[0] = s[0] * (1.0f / (float)BATCH);
}

/* ---------------- launch ---------------- */

extern "C" void kernel_launch(void* const* d_in, const int* in_sizes, int n_in,
                              void* d_out, int out_size, void* d_ws, size_t ws_size,
                              hipStream_t stream) {
  const float* v_feat   = (const float*)d_in[0];
  const float* id_emb   = (const float*)d_in[1];
  const float* W1       = (const float*)d_in[2];
  const float* b1       = (const float*)d_in[3];
  const float* W2       = (const float*)d_in[4];
  const float* b2       = (const float*)d_in[5];
  const int*   user_t   = (const int*)d_in[6];
  const int*   item_t   = (const int*)d_in[7];
  const int*   rand_idx = (const int*)d_in[8];
  float* out = (float*)d_out;
  char*  ws  = (char*)d_ws;

  /* workspace layout (bytes), all 256-aligned */
  float*     feature = (float*)(ws);                      /* 200000*64*4 = 51,200,000 */
  _Float16*  W1t     = (_Float16*)(ws + 51200000);        /* 65,536  */
  _Float16*  W2t     = (_Float16*)(ws + 51265536);        /* 32,768  */
  int*       mask    = (int*)(ws + 51298304);             /* 1,114,112 */
  float*     partial = (float*)(ws + 52412416);           /* 8,192   */

  convert_weights<<<192, 256, 0, stream>>>(W1, W2, W1t, W2t);
  zero_mask<<<(NTOT + 255) / 256, 256, 0, stream>>>(mask);
  set_mask<<<(NREPL + 255) / 256, 256, 0, stream>>>(rand_idx, mask);

  encoder_kernel<<<NUM_ITEM / 64, 128, LDS_BYTES, stream>>>(v_feat, W1t, W2t, b1, b2, feature);

  loss_kernel<<<BATCH / 8, 256, 0, stream>>>(feature, id_emb, user_t, item_t, mask, partial);
  final_reduce<<<1, 256, 0, stream>>>(partial, out);
}
